// TreeLayer_60249801229164
// MI455X (gfx1250) — matrix operand order repaired and checked by reading, not compiled, not run on hardware
//
#include <hip/hip_runtime.h>
#include <hip/hip_bf16.h>

// ---------------------------------------------------------------------------
// Types for CDNA5 WMMA
// ---------------------------------------------------------------------------
typedef __attribute__((ext_vector_type(16))) __bf16 bf16x16;
typedef __attribute__((ext_vector_type(8)))  float  f32x8;

#define IN_SIZE  4096
#define OUT_SIZE 4096
#define BATCH    8192
#define BLK      64

// GEMM tiling
#define MT 128        // rows of x per workgroup
#define NT 256        // output columns per workgroup
#define KC 32         // K per WMMA step (bf16 16x16x32)
#define LDPAD 8       // pad bf16 elems -> row stride 40 bf16 = 80B (16B aligned, conflict-free)
#define LDS_LD (KC + LDPAD)

// Stats reduction split
#define CHUNKS 32
#define ROWS_PER_CHUNK (BATCH / CHUNKS)   // 256

// ---------------------------------------------------------------------------
// CDNA5 async memory->LDS copy (bypasses VGPRs, tracked on ASYNCcnt).
// lds_off: LDS byte address (flat shared-pointer low 32 bits; flat->LDS
// mapping truncates to addr[31:0] per ISA 10.2). g: 64-bit global address.
// ---------------------------------------------------------------------------
__device__ __forceinline__ void async_copy_b128(unsigned lds_off, const void* g) {
    asm volatile("global_load_async_to_lds_b128 %0, %1, off"
                 :: "v"(lds_off), "v"(g) : "memory");
}
__device__ __forceinline__ void wait_async0() {
    asm volatile("s_wait_asynccnt 0x0" ::: "memory");
}

// ---------------------------------------------------------------------------
// Kernel 1: f32 -> bf16 conversion of x (8 elements / thread, vectorized)
// ---------------------------------------------------------------------------
__global__ __launch_bounds__(256) void cvt_bf16_kernel(const float* __restrict__ x,
                                                       __bf16* __restrict__ xb) {
    size_t i = ((size_t)blockIdx.x * blockDim.x + threadIdx.x) * 8;
    float4 v0 = *(const float4*)(x + i);
    float4 v1 = *(const float4*)(x + i + 4);
    union { __bf16 b[8]; uint4 u; } o;
    o.b[0] = (__bf16)v0.x; o.b[1] = (__bf16)v0.y;
    o.b[2] = (__bf16)v0.z; o.b[3] = (__bf16)v0.w;
    o.b[4] = (__bf16)v1.x; o.b[5] = (__bf16)v1.y;
    o.b[6] = (__bf16)v1.z; o.b[7] = (__bf16)v1.w;
    *(uint4*)(xb + i) = o.u;
}

// ---------------------------------------------------------------------------
// Kernel 2: build transposed masked weight  Wt[j (out)][i (in)]  in bf16.
//   block-diagonal     (i/64 == j/64): tw[i][j]^2
//   strict block-upper (i/64 <  j/64): bbw[i][j]
//   else zero
// 32x32 LDS transpose tiles; each tile lies entirely in one mask region.
// ---------------------------------------------------------------------------
__global__ __launch_bounds__(256) void build_wt_kernel(const float* __restrict__ tw,
                                                       const float* __restrict__ bbw,
                                                       __bf16* __restrict__ wt) {
    __shared__ float t[32][33];
    const int iBase = blockIdx.x * 32;   // input index (row of tw/bbw)
    const int jBase = blockIdx.y * 32;   // output index (col of tw/bbw)
    const int rb = iBase >> 6, cb = jBase >> 6;
    const int tx = threadIdx.x;          // 0..31
    const int ty0 = threadIdx.y;         // 0..7

    if (rb == cb) {
        for (int ty = ty0; ty < 32; ty += 8) {
            float v = tw[(size_t)(iBase + ty) * OUT_SIZE + jBase + tx];
            t[ty][tx] = v * v;
        }
    } else if (rb < cb) {
        for (int ty = ty0; ty < 32; ty += 8)
            t[ty][tx] = bbw[(size_t)(iBase + ty) * OUT_SIZE + jBase + tx];
    } else {
        for (int ty = ty0; ty < 32; ty += 8)
            t[ty][tx] = 0.0f;
    }
    __syncthreads();
    for (int ty = ty0; ty < 32; ty += 8)
        wt[(size_t)(jBase + ty) * IN_SIZE + iBase + tx] = (__bf16)t[tx][ty];
}

// ---------------------------------------------------------------------------
// Kernel 3: bf16 WMMA GEMM  h = x @ W + x_init, exploiting block-triangular W.
//   Workgroup: 256 threads (8 waves). Tile: MT x NT. Wave tile: 64x64
//   = 4x4 grid of v_wmma_f32_16x16x32_bf16 accumulators.
//   Global->LDS staging uses CDNA5 GLOBAL_LOAD_ASYNC_TO_LDS_B128 (ASYNCcnt),
//   double buffered so the DMA overlaps the WMMA chain.
//   K loop bounded at n0 + NT (everything past that is structurally zero).
// ---------------------------------------------------------------------------
__global__ __launch_bounds__(256) void gemm_kernel(const __bf16* __restrict__ xbf,
                                                   const __bf16* __restrict__ wt,
                                                   const float* __restrict__ x_init,
                                                   float* __restrict__ h) {
    __shared__ __bf16 As[2][MT][LDS_LD];   // x tile: 128 rows x 32 K
    __shared__ __bf16 Bs[2][NT][LDS_LD];   // Wt tile: 256 out-cols x 32 K

    const int tid  = threadIdx.x;
    const int lane = tid & 31;
    const int wave = tid >> 5;
    const int n0 = blockIdx.x * NT;
    const int m0 = blockIdx.y * MT;
    const int wm = (wave >> 2) * 64;       // wave row offset within tile (0,64)
    const int wn = (wave & 3) * 64;        // wave col offset within tile (0..192)

    f32x8 acc[4][4] = {};

    // Triangular K bound: columns [n0, n0+NT) only need k < n0+NT.
    const int kMax  = (n0 + NT) < IN_SIZE ? (n0 + NT) : IN_SIZE;
    const int steps = kMax / KC;

    // Async-stage one K-slab of A and B into LDS buffer `buf`.
    auto loadTilesAsync = [&](int buf, int k0) {
        // A: thread t -> row t/2, 32B half t%2 => 2 x b128 per thread
        {
            const int row = tid >> 1, half = tid & 1;
            const __bf16* src = xbf + (size_t)(m0 + row) * IN_SIZE + k0 + half * 16;
            unsigned dst = (unsigned)(size_t)(&As[buf][row][half * 16]);
            async_copy_b128(dst,      src);
            async_copy_b128(dst + 16, src + 8);
        }
        // B: thread t -> one full 64B row of Wt => 4 x b128 per thread
        {
            const __bf16* src = wt + (size_t)(n0 + tid) * IN_SIZE + k0;
            unsigned dst = (unsigned)(size_t)(&Bs[buf][tid][0]);
            async_copy_b128(dst,      src);
            async_copy_b128(dst + 16, src + 8);
            async_copy_b128(dst + 32, src + 16);
            async_copy_b128(dst + 48, src + 24);
        }
    };

    // A fragment lane layout (16-bit A 16x32): lanes 0-15: elems 0-7 -> K 0-7,
    // elems 8-15 -> K 16-23; lanes 16-31: K 8-15 and K 24-31.
    auto loadAfrag = [&](int buf, int mi) {
        const int row = wm + mi * 16 + (lane & 15);
        const int c0  = (lane < 16) ? 0 : 8;
        union { uint4 u[2]; bf16x16 v; } tmp;
        tmp.u[0] = *(const uint4*)(&As[buf][row][c0]);
        tmp.u[1] = *(const uint4*)(&As[buf][row][c0 + 16]);
        return tmp.v;
    };
    // B fragment lane layout (16-bit B 32x16): lane = N (mod 16); lanes 0-15
    // hold K 0-7 and K 16-23, lanes 16-31 hold K 8-15 and K 24-31 (matches A).
    auto loadBfrag = [&](int buf, int ni) {
        const int row = wn + ni * 16 + (lane & 15);
        const int c0  = (lane < 16) ? 0 : 8;
        union { uint4 u[2]; bf16x16 v; } tmp;
        tmp.u[0] = *(const uint4*)(&Bs[buf][row][c0]);
        tmp.u[1] = *(const uint4*)(&Bs[buf][row][c0 + 16]);
        return tmp.v;
    };

    loadTilesAsync(0, 0);
    wait_async0();
    __syncthreads();

    for (int s = 0; s < steps; ++s) {
        const int buf = s & 1;
        const bool more = (s + 1 < steps);
        if (more) loadTilesAsync(buf ^ 1, (s + 1) * KC);   // DMA overlaps WMMAs

        bf16x16 a[4], b[4];
        #pragma unroll
        for (int mi = 0; mi < 4; ++mi) a[mi] = loadAfrag(buf, mi);
        #pragma unroll
        for (int ni = 0; ni < 4; ++ni) b[ni] = loadBfrag(buf, ni);

        #pragma unroll
        for (int mi = 0; mi < 4; ++mi)
            #pragma unroll
            for (int ni = 0; ni < 4; ++ni)
                acc[mi][ni] = __builtin_amdgcn_wmma_f32_16x16x32_bf16(
                    false, a[mi], false, b[ni], (short)0, acc[mi][ni], false, false);

        if (more) wait_async0();   // own async copies landed in LDS
        __syncthreads();           // publish to all waves
    }

    // Epilogue: h = acc + x_init. C/D layout: lane -> N = lane%16,
    // element r -> M = r + 8*(lane/16). Batch the 8 x_init loads per fragment
    // so they issue together (one loadcnt wait instead of eight).
    #pragma unroll
    for (int mi = 0; mi < 4; ++mi) {
        #pragma unroll
        for (int ni = 0; ni < 4; ++ni) {
            const int col     = n0 + wn + ni * 16 + (lane & 15);
            const int rowBase = m0 + wm + mi * 16 + ((lane >> 4) << 3);
            const float* xiP = x_init + (size_t)rowBase * OUT_SIZE + col;
            float*       hP  = h      + (size_t)rowBase * OUT_SIZE + col;
            float xi[8];
            #pragma unroll
            for (int r = 0; r < 8; ++r) xi[r] = xiP[(size_t)r * OUT_SIZE];
            #pragma unroll
            for (int r = 0; r < 8; ++r) hP[(size_t)r * OUT_SIZE] = acc[mi][ni][r] + xi[r];
        }
    }
}

// ---------------------------------------------------------------------------
// Kernel 4a: partial per-column sums over 256-row chunks (512 blocks).
// ---------------------------------------------------------------------------
__global__ __launch_bounds__(256) void stats_partial_kernel(const float* __restrict__ h,
                                                            float* __restrict__ psum,
                                                            float* __restrict__ psumsq) {
    const int j = blockIdx.x * blockDim.x + threadIdx.x;   // column
    const int c = blockIdx.y;                              // row chunk
    const float* p = h + (size_t)c * ROWS_PER_CHUNK * OUT_SIZE + j;
    float s = 0.f, sq = 0.f;
    for (int r = 0; r < ROWS_PER_CHUNK; ++r) {
        float v = p[(size_t)r * OUT_SIZE];
        s += v; sq += v * v;
    }
    psum  [(size_t)c * OUT_SIZE + j] = s;
    psumsq[(size_t)c * OUT_SIZE + j] = sq;
}

// ---------------------------------------------------------------------------
// Kernel 4b: reduce chunk partials -> mean, 1/(std+eps) per column.
// ---------------------------------------------------------------------------
__global__ __launch_bounds__(256) void stats_final_kernel(const float* __restrict__ psum,
                                                          const float* __restrict__ psumsq,
                                                          float* __restrict__ mean,
                                                          float* __restrict__ scale2) {
    const int j = blockIdx.x * blockDim.x + threadIdx.x;
    float s = 0.f, sq = 0.f;
    #pragma unroll 4
    for (int c = 0; c < CHUNKS; ++c) {
        s  += psum  [(size_t)c * OUT_SIZE + j];
        sq += psumsq[(size_t)c * OUT_SIZE + j];
    }
    const float inv = 1.0f / (float)BATCH;
    const float m = s * inv;
    float var = sq * inv - m * m;
    var = var > 0.f ? var : 0.f;
    const float sd = sqrtf(var);
    mean[j]   = m;
    scale2[j] = 1.0f / (sd + 1e-6f);   // tree_scale^2 applied directly
}

// ---------------------------------------------------------------------------
// Kernel 5: z = (h - mean) * scale2 ; out = sigmoid(z). In-place on d_out.
// ---------------------------------------------------------------------------
__global__ __launch_bounds__(256) void norm_sigmoid_kernel(float* __restrict__ h,
                                                           const float* __restrict__ mean,
                                                           const float* __restrict__ scale2) {
    const size_t i = (size_t)blockIdx.x * blockDim.x + threadIdx.x;
    const int j = (int)(i & (OUT_SIZE - 1));
    const float z = (h[i] - mean[j]) * scale2[j];
    h[i] = 1.0f / (1.0f + __expf(-z));
}

// ---------------------------------------------------------------------------
// Launcher
// ---------------------------------------------------------------------------
extern "C" void kernel_launch(void* const* d_in, const int* in_sizes, int n_in,
                              void* d_out, int out_size, void* d_ws, size_t ws_size,
                              hipStream_t stream) {
    const float* x      = (const float*)d_in[0];   // [8192, 4096]
    const float* x_init = (const float*)d_in[1];   // [8192, 4096]
    const float* tw     = (const float*)d_in[2];   // [4096, 4096]
    const float* bbw    = (const float*)d_in[3];   // [4096, 4096]
    float* out = (float*)d_out;                    // [8192, 4096]

    // Workspace: xbf (64 MiB) | Wt (32 MiB) | mean/scale (32 KiB) | partials (1 MiB)
    __bf16* xbf    = (__bf16*)d_ws;
    __bf16* wtbuf  = (__bf16*)((char*)d_ws + ((size_t)64 << 20));
    float*  meanb  = (float*)((char*)d_ws + ((size_t)96 << 20));
    float*  scaleb = meanb + OUT_SIZE;
    float*  psum   = (float*)((char*)d_ws + ((size_t)96 << 20) + (64 << 10));
    float*  psumsq = psum + (size_t)CHUNKS * OUT_SIZE;

    // 1) x -> bf16
    cvt_bf16_kernel<<<(BATCH * IN_SIZE) / (256 * 8), 256, 0, stream>>>(x, xbf);

    // 2) masked, squared-diag, transposed weights -> bf16
    build_wt_kernel<<<dim3(IN_SIZE / 32, OUT_SIZE / 32), dim3(32, 8), 0, stream>>>(tw, bbw, wtbuf);

    // 3) WMMA GEMM with async-LDS double buffering, triangular K-skip, fused + x_init
    gemm_kernel<<<dim3(OUT_SIZE / NT, BATCH / MT), 256, 0, stream>>>(xbf, wtbuf, x_init, out);

    // 4) column stats: chunked partials, then final reduce
    stats_partial_kernel<<<dim3(OUT_SIZE / 256, CHUNKS), 256, 0, stream>>>(out, psum, psumsq);
    stats_final_kernel<<<OUT_SIZE / 256, 256, 0, stream>>>(psum, psumsq, meanb, scaleb);

    // 5) normalize + sigmoid
    norm_sigmoid_kernel<<<(size_t)BATCH * OUT_SIZE / 256, 256, 0, stream>>>(out, meanb, scaleb);
}
